// Decoder_51702816309853
// MI455X (gfx1250) — matrix operand order repaired and checked by reading, not compiled, not run on hardware
//
#include <hip/hip_runtime.h>
#include <hip/hip_bf16.h>
#include <math.h>

// ----- problem constants -----
#define BB    64      // batch
#define TENC  512
#define ENCD  512
#define TDEC  600
#define NMEL  80
#define RNN_N 1024
#define NATT  128
#define CCH   32
#define KSZ_N 31
#define NPRE  256

typedef unsigned short u16;
typedef unsigned int   u32;
typedef __attribute__((ext_vector_type(16))) __bf16 v16bf;
typedef __attribute__((ext_vector_type(8)))  float  v8f;
typedef __attribute__((ext_vector_type(4)))  u32    u32x4;
typedef __attribute__((ext_vector_type(4)))  float  f32x4;

union FragBF { v16bf v; u16 s[16]; u32x4 q[2]; };

__device__ __forceinline__ u16 f2bf(float f) {
  u32 u = __builtin_bit_cast(u32, f);
  return (u16)((u + 0x7FFFu + ((u >> 16) & 1u)) >> 16);   // RNE
}

__device__ __forceinline__ void chunk_f32(u16* d, const float* p) {
  f32x4 a = *(const f32x4*)p;
  f32x4 b = *(const f32x4*)(p + 4);
  d[0]=f2bf(a[0]); d[1]=f2bf(a[1]); d[2]=f2bf(a[2]); d[3]=f2bf(a[3]);
  d[4]=f2bf(b[0]); d[5]=f2bf(b[1]); d[6]=f2bf(b[2]); d[7]=f2bf(b[3]);
}

// B stored transposed (N x K) bf16: per-lane two 16B chunks (k0+8h, k0+16+8h).
__device__ __forceinline__ v16bf bfragT(const u16* BT, int ldk, int n0, int k0, int lane) {
  FragBF f;
  const u16* p = BT + (size_t)(n0 + (lane & 15)) * ldk + k0 + 8 * (lane >> 4);
  f.q[0] = *(const u32x4*)p;
  f.q[1] = *(const u32x4*)(p + 16);
  return f.v;
}

// A (M x K) bf16 row-major: two 16B chunks per lane.
__device__ __forceinline__ v16bf afrag_bf(const u16* A, int ldk, int m0, int k0, int lane) {
  FragBF f;
  const u16* p = A + (size_t)(m0 + (lane & 15)) * ldk + k0 + 8 * (lane >> 4);
  f.q[0] = *(const u32x4*)p;
  f.q[1] = *(const u32x4*)(p + 16);
  return f.v;
}

__device__ __forceinline__ v8f wmma_bf16(v16bf a, v16bf b, v8f c) {
  return __builtin_amdgcn_wmma_f32_16x16x32_bf16(false, a, false, b, (short)0, c, false, false);
}

__device__ __forceinline__ float sigf(float x) { return 1.0f / (1.0f + __expf(-x)); }

#define V8F_ZERO {0.f,0.f,0.f,0.f,0.f,0.f,0.f,0.f}

// =====================  one-time setup kernels  =====================

// W1cat^T (4096 x 1792): K order = [prenet 256 | ctx 512 | h1 1024]
__global__ __launch_bounds__(256) void k_pack_w1(const float* __restrict__ Wih1,
                                                 const float* __restrict__ Whh1,
                                                 u16* __restrict__ W1) {
  int idx = blockIdx.x * 256 + threadIdx.x;
  if (idx >= 4096 * 1792) return;
  int n = idx / 1792, k = idx % 1792;
  float v = (k < 768) ? Wih1[(size_t)k * 4096 + n] : Whh1[(size_t)(k - 768) * 4096 + n];
  W1[idx] = f2bf(v);
}

// W2cat^T (4096 x 2560): K order = [h1 1024 | ctx 512 | h2 1024]
__global__ __launch_bounds__(256) void k_pack_w2(const float* __restrict__ Wih2,
                                                 const float* __restrict__ Whh2,
                                                 u16* __restrict__ W2) {
  int idx = blockIdx.x * 256 + threadIdx.x;
  if (idx >= 4096 * 2560) return;
  int n = idx / 2560, k = idx % 2560;
  float v = (k < 1536) ? Wih2[(size_t)k * 4096 + n] : Whh2[(size_t)(k - 1536) * 4096 + n];
  W2[idx] = f2bf(v);
}

// projcat^T (96 x 1536): rows 0..79 = proj_W cols, row 80 = gate_W, rest zero
__global__ __launch_bounds__(256) void k_pack_prj(const float* __restrict__ projW,
                                                  const float* __restrict__ gateW,
                                                  u16* __restrict__ PRJ) {
  int idx = blockIdx.x * 256 + threadIdx.x;
  if (idx >= 96 * 1536) return;
  int n = idx / 1536, k = idx % 1536;
  float v = (n < 80) ? projW[(size_t)k * 80 + n] : (n == 80 ? gateW[k] : 0.f);
  PRJ[idx] = f2bf(v);
}

// generic (K x N) f32 -> (N x K) bf16 transpose
__global__ __launch_bounds__(256) void k_pack_t(const float* __restrict__ src,
                                                u16* __restrict__ dst, int N, int K) {
  int idx = blockIdx.x * 256 + threadIdx.x;
  if (idx >= N * K) return;
  int n = idx / K, k = idx % K;
  dst[idx] = f2bf(src[(size_t)k * N + n]);
}

// processed_memory = memory @ mem_W  (M=B*TENC=32768, K=512, N=128), f32 out
__global__ __launch_bounds__(256) void k_pm(const float* __restrict__ memory,
                                            const u16* __restrict__ MWT,
                                            float* __restrict__ pm) {
  int wave = threadIdx.x >> 5, lane = threadIdx.x & 31;
  int tile = blockIdx.x * 8 + wave;          // 2048 mtiles x 8 ntiles
  int m0 = (tile >> 3) * 16, n0 = (tile & 7) * 16;
  int half = lane >> 4;
  const float* row = memory + (size_t)(m0 + (lane & 15)) * ENCD;
  v8f acc = V8F_ZERO;
  for (int ks = 0; ks < 16; ++ks) {
    int k0 = ks * 32;
    FragBF fa;
    chunk_f32(fa.s,     row + k0 + 8 * half);
    chunk_f32(fa.s + 8, row + k0 + 16 + 8 * half);
    v16bf fb = bfragT(MWT, 512, n0, k0, lane);
    acc = wmma_bf16(fa.v, fb, acc);
  }
  int n = n0 + (lane & 15);
  #pragma unroll
  for (int r = 0; r < 8; ++r)
    pm[(size_t)(m0 + r + 8 * half) * NATT + n] = acc[r];
}

// prenet layer 1 (VALU, runs once): row = t*64+b, x_t = targets[:, :, t-1] (0 at t=0)
__global__ __launch_bounds__(256) void k_pre1(const float* __restrict__ tgt,
                                              const float* __restrict__ W0,
                                              u16* __restrict__ tmp) {
  int idx = blockIdx.x * 256 + threadIdx.x;
  if (idx >= TDEC * BB * NPRE) return;
  int row = idx >> 8, n = idx & 255;
  int t = row >> 6, b = row & 63;
  float acc = 0.f;
  if (t > 0) {
    const float* x = tgt + (size_t)b * NMEL * TDEC + (t - 1);
    for (int k = 0; k < NMEL; ++k) acc += x[(size_t)k * TDEC] * W0[k * NPRE + n];
  }
  tmp[idx] = f2bf(fmaxf(acc, 0.f));
}

// prenet layer 2 (WMMA): Dall[t*64+b][256] = relu(tmp @ pre_W1)
__global__ __launch_bounds__(256) void k_pre2(const u16* __restrict__ tmp,
                                              const u16* __restrict__ PW1,
                                              u16* __restrict__ Dall) {
  int wave = threadIdx.x >> 5, lane = threadIdx.x & 31;
  int tile = blockIdx.x * 8 + wave;          // 2400 mtiles x 16 ntiles
  int m0 = (tile >> 4) * 16, n0 = (tile & 15) * 16;
  v8f acc = V8F_ZERO;
  for (int ks = 0; ks < 8; ++ks) {
    int k0 = ks * 32;
    v16bf fa = afrag_bf(tmp, NPRE, m0, k0, lane);
    v16bf fb = bfragT(PW1, NPRE, n0, k0, lane);
    acc = wmma_bf16(fa, fb, acc);
  }
  int half = lane >> 4, n = n0 + (lane & 15);
  #pragma unroll
  for (int r = 0; r < 8; ++r)
    Dall[(size_t)(m0 + r + 8 * half) * NPRE + n] = f2bf(fmaxf(acc[r], 0.f));
}

// =====================  per-step kernels  =====================
// All recurrent GEMM operands (Dall, ctxb, h1b, h2b) are kept as bf16 mirrors
// by their producers, so A-fragment loads are pure 16B loads (no VALU cvt).

__device__ __forceinline__ void chunk_k1(u16* d, int m, int k, int t,
    const u16* __restrict__ Dall, const u16* __restrict__ ctxb,
    const u16* __restrict__ h1b) {
  const u16* p;
  if (k < NPRE)             p = Dall + ((size_t)t * BB + m) * NPRE + k;
  else if (k < NPRE + ENCD) p = ctxb + (size_t)m * ENCD + (k - NPRE);
  else                      p = h1b + (size_t)m * RNN_N + (k - NPRE - ENCD);
  *(u32x4*)d = *(const u32x4*)p;
}

__device__ __forceinline__ void chunk_k3(u16* d, int m, int k,
    const u16* __restrict__ h1b, const u16* __restrict__ ctxb,
    const u16* __restrict__ h2b) {
  const u16* p;
  if (k < RNN_N)             p = h1b + (size_t)m * RNN_N + k;
  else if (k < RNN_N + ENCD) p = ctxb + (size_t)m * ENCD + (k - RNN_N);
  else                       p = h2b + (size_t)m * RNN_N + (k - RNN_N - ENCD);
  *(u32x4*)d = *(const u32x4*)p;
}

__device__ __forceinline__ void chunk_k4(u16* d, int m, int k,
    const u16* __restrict__ h2b, const u16* __restrict__ ctxb) {
  const u16* p;
  if (k < RNN_N) p = h2b + (size_t)m * RNN_N + k;
  else           p = ctxb + (size_t)m * ENCD + (k - RNN_N);
  *(u32x4*)d = *(const u32x4*)p;
}

// LSTM1: gates = [D_t | ctx | h1] @ W1cat + b  -> h1' (f32 + bf16), c1'.
// Each wave owns all 4 gates (i,f,g,o) of its 32-col hidden slice ->
// nonlinearity entirely in registers, no inter-wave exchange.
__global__ __launch_bounds__(256) void k_lstm1(
    const u16* __restrict__ Dall, const u16* __restrict__ W1,
    const u16* __restrict__ ctxb_c, const u16* __restrict__ h1b_c,
    const float* __restrict__ c1_c, const float* __restrict__ bih,
    const float* __restrict__ bhh, float* __restrict__ h1_n,
    u16* __restrict__ h1b_n, float* __restrict__ c1_n, int t) {
  const int wave = threadIdx.x >> 5, lane = threadIdx.x & 31;
  const int m0  = 16 * (wave & 3);
  const int ncb = blockIdx.x * 64 + 32 * (wave >> 2);
  const int m = m0 + (lane & 15), half = lane >> 4;
  v8f acc[4][2];
  #pragma unroll
  for (int g = 0; g < 4; ++g) { v8f z = V8F_ZERO; acc[g][0] = z; acc[g][1] = z; }
  for (int ks = 0; ks < 56; ++ks) {
    const int k0 = ks * 32;
    FragBF fa;
    chunk_k1(fa.s,     m, k0 + 8 * half,      t, Dall, ctxb_c, h1b_c);
    chunk_k1(fa.s + 8, m, k0 + 16 + 8 * half, t, Dall, ctxb_c, h1b_c);
    if (ks + 1 < 56)
      __builtin_prefetch(W1 + (size_t)ncb * 1792 + k0 + 32, 0, 0);
    #pragma unroll
    for (int g = 0; g < 4; ++g)
      #pragma unroll
      for (int nn = 0; nn < 2; ++nn) {
        v16bf fb = bfragT(W1, 1792, g * 1024 + ncb + 16 * nn, k0, lane);
        acc[g][nn] = wmma_bf16(fa.v, fb, acc[g][nn]);
      }
  }
  #pragma unroll
  for (int nn = 0; nn < 2; ++nn) {
    int j = ncb + 16 * nn + (lane & 15);
    float bi = bih[j]        + bhh[j];
    float bf = bih[1024 + j] + bhh[1024 + j];
    float bg = bih[2048 + j] + bhh[2048 + j];
    float bo = bih[3072 + j] + bhh[3072 + j];
    #pragma unroll
    for (int r = 0; r < 8; ++r) {
      int mm = m0 + r + 8 * half;
      float ii = acc[0][nn][r] + bi, ff = acc[1][nn][r] + bf;
      float gg = acc[2][nn][r] + bg, oo = acc[3][nn][r] + bo;
      float c  = sigf(ff) * c1_c[(size_t)mm * RNN_N + j] + sigf(ii) * tanhf(gg);
      float h  = sigf(oo) * tanhf(c);
      h1_n[(size_t)mm * RNN_N + j]  = h;
      h1b_n[(size_t)mm * RNN_N + j] = f2bf(h);
      c1_n[(size_t)mm * RNN_N + j]  = c;
    }
  }
}

// LSTM2: gates = [h1' | ctx' | h2] @ W2cat + b -> h2' (bf16 only), c2'
__global__ __launch_bounds__(256) void k_lstm2(
    const u16* __restrict__ W2, const u16* __restrict__ h1b_n,
    const u16* __restrict__ ctxb_n, const u16* __restrict__ h2b_c,
    const float* __restrict__ c2_c, const float* __restrict__ bih,
    const float* __restrict__ bhh, u16* __restrict__ h2b_n,
    float* __restrict__ c2_n) {
  const int wave = threadIdx.x >> 5, lane = threadIdx.x & 31;
  const int m0  = 16 * (wave & 3);
  const int ncb = blockIdx.x * 64 + 32 * (wave >> 2);
  const int m = m0 + (lane & 15), half = lane >> 4;
  v8f acc[4][2];
  #pragma unroll
  for (int g = 0; g < 4; ++g) { v8f z = V8F_ZERO; acc[g][0] = z; acc[g][1] = z; }
  for (int ks = 0; ks < 80; ++ks) {
    const int k0 = ks * 32;
    FragBF fa;
    chunk_k3(fa.s,     m, k0 + 8 * half,      h1b_n, ctxb_n, h2b_c);
    chunk_k3(fa.s + 8, m, k0 + 16 + 8 * half, h1b_n, ctxb_n, h2b_c);
    if (ks + 1 < 80)
      __builtin_prefetch(W2 + (size_t)ncb * 2560 + k0 + 32, 0, 0);
    #pragma unroll
    for (int g = 0; g < 4; ++g)
      #pragma unroll
      for (int nn = 0; nn < 2; ++nn) {
        v16bf fb = bfragT(W2, 2560, g * 1024 + ncb + 16 * nn, k0, lane);
        acc[g][nn] = wmma_bf16(fa.v, fb, acc[g][nn]);
      }
  }
  #pragma unroll
  for (int nn = 0; nn < 2; ++nn) {
    int j = ncb + 16 * nn + (lane & 15);
    float bi = bih[j]        + bhh[j];
    float bf = bih[1024 + j] + bhh[1024 + j];
    float bg = bih[2048 + j] + bhh[2048 + j];
    float bo = bih[3072 + j] + bhh[3072 + j];
    #pragma unroll
    for (int r = 0; r < 8; ++r) {
      int mm = m0 + r + 8 * half;
      float ii = acc[0][nn][r] + bi, ff = acc[1][nn][r] + bf;
      float gg = acc[2][nn][r] + bg, oo = acc[3][nn][r] + bo;
      float c  = sigf(ff) * c2_c[(size_t)mm * RNN_N + j] + sigf(ii) * tanhf(gg);
      h2b_n[(size_t)mm * RNN_N + j] = f2bf(sigf(oo) * tanhf(c));
      c2_n[(size_t)mm * RNN_N + j]  = c;
    }
  }
}

// location-sensitive attention: one block per batch, 512 threads (one per t_enc)
__global__ __launch_bounds__(512) void k_attn(
    const float* __restrict__ h1, const float* __restrict__ qW,
    const float* __restrict__ locW, const float* __restrict__ convW,
    const float* __restrict__ wW, const float* __restrict__ wb,
    const float* __restrict__ pm, const float* __restrict__ memory,
    const int* __restrict__ mlen, const float* __restrict__ cum_c,
    float* __restrict__ cum_n, u16* __restrict__ ctxb_n,
    float* __restrict__ align_out, int t) {
  __shared__ float s_cum[TENC];
  __shared__ float s_pq[NATT];
  __shared__ float s_locW[CCH * NATT];
  __shared__ float s_convW[CCH * KSZ_N];
  __shared__ float s_e[TENC];
  __shared__ float s_red[16];
  __shared__ float s_scalar[2];
  const int b = blockIdx.x, tid = threadIdx.x;

  s_cum[tid] = cum_c[b * TENC + tid];
  for (int i = tid; i < CCH * NATT;  i += 512) s_locW[i]  = locW[i];
  for (int i = tid; i < CCH * KSZ_N; i += 512) s_convW[i] = convW[i];
  if (tid < NATT) {
    float acc = 0.f;
    const float* h = h1 + (size_t)b * RNN_N;
    for (int k = 0; k < RNN_N; ++k) acc += h[k] * qW[k * NATT + tid];
    s_pq[tid] = acc;
  }
  __syncthreads();

  const int len = mlen[b];
  {
    float loc[CCH];
    #pragma unroll
    for (int c = 0; c < CCH; ++c) {
      float s = 0.f;
      #pragma unroll
      for (int k = 0; k < KSZ_N; ++k) {
        int tt = tid + k - 15;
        float v = (tt >= 0 && tt < TENC) ? s_cum[tt] : 0.f;
        s += s_convW[c * KSZ_N + k] * v;
      }
      loc[c] = s;
    }
    const float* pmrow = pm + ((size_t)b * TENC + tid) * NATT;
    float e = 0.f;
    for (int a = 0; a < NATT; ++a) {
      float paw = 0.f;
      #pragma unroll
      for (int c = 0; c < CCH; ++c) paw += loc[c] * s_locW[c * NATT + a];
      e += wW[a] * tanhf(s_pq[a] + paw + pmrow[a]);
    }
    e += wb[0];
    if (tid >= len) e = -1e9f;
    s_e[tid] = e;
  }
  __syncthreads();

  // softmax over 512 (16 waves, wave32 shuffles + LDS)
  float e = s_e[tid], mx = e;
  for (int off = 16; off; off >>= 1) mx = fmaxf(mx, __shfl_xor(mx, off, 32));
  if ((tid & 31) == 0) s_red[tid >> 5] = mx;
  __syncthreads();
  if (tid == 0) { float m = -1e30f; for (int i = 0; i < 16; ++i) m = fmaxf(m, s_red[i]); s_scalar[0] = m; }
  __syncthreads();
  float ex = __expf(e - s_scalar[0]);
  float sm = ex;
  for (int off = 16; off; off >>= 1) sm += __shfl_xor(sm, off, 32);
  __syncthreads();                       // s_red reads done before rewrite
  if ((tid & 31) == 0) s_red[tid >> 5] = sm;
  __syncthreads();
  if (tid == 0) { float s = 0.f; for (int i = 0; i < 16; ++i) s += s_red[i]; s_scalar[1] = s; }
  __syncthreads();

  float attn = ex / s_scalar[1];
  s_e[tid] = attn;                       // reuse as attn
  align_out[(size_t)b * TDEC * TENC + (size_t)t * TENC + tid] = attn;
  cum_n[b * TENC + tid] = s_cum[tid] + attn;
  __syncthreads();

  // context: thread = d-index, coalesced over d each iteration; bf16 out
  float acc = 0.f;
  const float* mb = memory + (size_t)b * TENC * ENCD;
  for (int tt = 0; tt < TENC; ++tt) acc += s_e[tt] * mb[(size_t)tt * ENCD + tid];
  ctxb_n[b * ENCD + tid] = f2bf(acc);
}

// projection: [h2' | ctx'] @ projcat (96 cols: 80 mel + 1 gate + pad). 1 block.
__global__ __launch_bounds__(256) void k_proj(
    const u16* __restrict__ PRJ, const u16* __restrict__ h2b_n,
    const u16* __restrict__ ctxb_n, const float* __restrict__ gateB,
    float* __restrict__ out_mel, float* __restrict__ out_gate, int t) {
  const int wave = threadIdx.x >> 5, lane = threadIdx.x & 31;
  const int m0 = 16 * (wave & 3);
  const int ng = (wave >> 2) * 3;
  const int m = m0 + (lane & 15), half = lane >> 4;
  v8f acc[3];
  { v8f z = V8F_ZERO; acc[0] = z; acc[1] = z; acc[2] = z; }
  for (int ks = 0; ks < 48; ++ks) {
    const int k0 = ks * 32;
    FragBF fa;
    chunk_k4(fa.s,     m, k0 + 8 * half,      h2b_n, ctxb_n);
    chunk_k4(fa.s + 8, m, k0 + 16 + 8 * half, h2b_n, ctxb_n);
    #pragma unroll
    for (int j = 0; j < 3; ++j) {
      v16bf fb = bfragT(PRJ, 1536, (ng + j) * 16, k0, lane);
      acc[j] = wmma_bf16(fa.v, fb, acc[j]);
    }
  }
  #pragma unroll
  for (int j = 0; j < 3; ++j) {
    int n = (ng + j) * 16 + (lane & 15);
    #pragma unroll
    for (int r = 0; r < 8; ++r) {
      int mm = m0 + r + 8 * half;
      float v = acc[j][r];
      if (n < NMEL)       out_mel[(size_t)mm * NMEL * TDEC + (size_t)n * TDEC + t] = v;
      else if (n == NMEL) out_gate[(size_t)mm * TDEC + t] = v + gateB[0];
    }
  }
}

// =====================  host orchestration  =====================

extern "C" void kernel_launch(void* const* d_in, const int* in_sizes, int n_in,
                              void* d_out, int out_size, void* d_ws, size_t ws_size,
                              hipStream_t stream) {
  (void)in_sizes; (void)n_in; (void)out_size; (void)ws_size;
  const float* memory  = (const float*)d_in[0];
  const int*   mlen    = (const int*)d_in[1];
  const float* targets = (const float*)d_in[2];
  const float* qW      = (const float*)d_in[3];
  const float* memW    = (const float*)d_in[4];
  const float* locW    = (const float*)d_in[5];
  const float* convW   = (const float*)d_in[6];
  const float* wW      = (const float*)d_in[7];
  const float* wb      = (const float*)d_in[8];
  const float* preW0   = (const float*)d_in[9];
  const float* preW1   = (const float*)d_in[10];
  const float* Wih1    = (const float*)d_in[11];
  const float* Whh1    = (const float*)d_in[12];
  const float* bih1    = (const float*)d_in[13];
  const float* bhh1    = (const float*)d_in[14];
  const float* Wih2    = (const float*)d_in[15];
  const float* Whh2    = (const float*)d_in[16];
  const float* bih2    = (const float*)d_in[17];
  const float* bhh2    = (const float*)d_in[18];
  const float* projW   = (const float*)d_in[19];
  const float* gateW   = (const float*)d_in[20];
  const float* gateB   = (const float*)d_in[21];

  char* ws = (char*)d_ws;
  size_t off = 0;
  auto alloc = [&](size_t bytes) {
    size_t r = off; off = (off + bytes + 255) & ~(size_t)255; return r;
  };
  u16*   W1   = (u16*)(ws + alloc((size_t)4096 * 1792 * 2));
  u16*   W2   = (u16*)(ws + alloc((size_t)4096 * 2560 * 2));
  u16*   PRJ  = (u16*)(ws + alloc((size_t)96 * 1536 * 2));
  u16*   PW1  = (u16*)(ws + alloc((size_t)NPRE * NPRE * 2));
  u16*   MWT  = (u16*)(ws + alloc((size_t)NATT * ENCD * 2));
  float* PM   = (float*)(ws + alloc((size_t)BB * TENC * NATT * 4));
  u16*   TMP  = (u16*)(ws + alloc((size_t)TDEC * BB * NPRE * 2));
  u16*   DALL = (u16*)(ws + alloc((size_t)TDEC * BB * NPRE * 2));
  size_t stateOff = off;
  float* H1  = (float*)(ws + alloc(2 * (size_t)BB * RNN_N * 4));  // f32 (attention query)
  float* C1  = (float*)(ws + alloc(2 * (size_t)BB * RNN_N * 4));
  float* C2  = (float*)(ws + alloc(2 * (size_t)BB * RNN_N * 4));
  float* CUM = (float*)(ws + alloc(2 * (size_t)BB * TENC * 4));
  u16*   H1B = (u16*)(ws + alloc(2 * (size_t)BB * RNN_N * 2));    // bf16 mirrors
  u16*   H2B = (u16*)(ws + alloc(2 * (size_t)BB * RNN_N * 2));
  u16*   CTXB= (u16*)(ws + alloc(2 * (size_t)BB * ENCD * 2));
  size_t stateBytes = off - stateOff;

  float* out_mel  = (float*)d_out;
  float* out_gate = out_mel + (size_t)BB * NMEL * TDEC;
  float* out_algn = out_gate + (size_t)BB * TDEC;

  hipMemsetAsync(ws + stateOff, 0, stateBytes, stream);

  // one-time: pack weights to bf16 (transposed N x K), hoisted GEMMs
  k_pack_w1<<<(4096 * 1792 + 255) / 256, 256, 0, stream>>>(Wih1, Whh1, W1);
  k_pack_w2<<<(4096 * 2560 + 255) / 256, 256, 0, stream>>>(Wih2, Whh2, W2);
  k_pack_prj<<<(96 * 1536 + 255) / 256, 256, 0, stream>>>(projW, gateW, PRJ);
  k_pack_t<<<(NPRE * NPRE + 255) / 256, 256, 0, stream>>>(preW1, PW1, NPRE, NPRE);
  k_pack_t<<<(NATT * ENCD + 255) / 256, 256, 0, stream>>>(memW, MWT, NATT, ENCD);
  k_pm<<<2048, 256, 0, stream>>>(memory, MWT, PM);
  k_pre1<<<(TDEC * BB * NPRE + 255) / 256, 256, 0, stream>>>(targets, preW0, TMP);
  k_pre2<<<4800, 256, 0, stream>>>(TMP, PW1, DALL);

  const size_t hs = (size_t)BB * RNN_N, cs = (size_t)BB * ENCD, us = (size_t)BB * TENC;
  for (int t = 0; t < TDEC; ++t) {
    const int cur = t & 1, nxt = cur ^ 1;
    k_lstm1<<<16, 256, 0, stream>>>(DALL, W1, CTXB + cur * cs, H1B + cur * hs,
                                    C1 + cur * hs, bih1, bhh1,
                                    H1 + nxt * hs, H1B + nxt * hs, C1 + nxt * hs, t);
    k_attn<<<BB, 512, 0, stream>>>(H1 + nxt * hs, qW, locW, convW, wW, wb, PM,
                                   memory, mlen, CUM + cur * us, CUM + nxt * us,
                                   CTXB + nxt * cs, out_algn, t);
    k_lstm2<<<16, 256, 0, stream>>>(W2, H1B + nxt * hs, CTXB + nxt * cs,
                                    H2B + cur * hs, C2 + cur * hs, bih2, bhh2,
                                    H2B + nxt * hs, C2 + nxt * hs);
    k_proj<<<1, 256, 0, stream>>>(PRJ, H2B + nxt * hs, CTXB + nxt * cs, gateB,
                                  out_mel, out_gate, t);
  }
}
